// SimpleMACELayerFused_33509334843738
// MI455X (gfx1250) — matrix operand order repaired and checked by reading, not compiled, not run on hardware
//
#include <hip/hip_runtime.h>
#include <hip/hip_bf16.h>

// ---------------------------------------------------------------------------
// MACE-style fused layer for MI455X (gfx1250, wave32, WMMA).
//
// out[t,:] = b + sum_{e: tgt(e)=t} (feat[src(e)] (x) Y_l3(edge_vec(e))).flatten() @ W
//
// Restructured so the 512-wide contraction with W happens per edge-tile via
// v_wmma_f32_16x16x32_f16 (K=512 -> 16 WMMA steps x 2 N-blocks), so only 32
// f32 atomics per edge hit L2 (25.6M total) instead of 512 (410M in the
// reference's segment-sum structure). All reused state (node_features, W,
// out) is L2-resident on MI455X's 192MB L2; the streamed edge data is ~22MB.
// ---------------------------------------------------------------------------

typedef __attribute__((ext_vector_type(16))) _Float16 v16h;
typedef __attribute__((ext_vector_type(8)))  float    v8f;

#define HIDDEN 32
#define IRR    16
#define KDIM   (HIDDEN * IRR)   // 512
#define OUTC   32
#define WAVES  4
#define TILE_E 16
#define BLOCK_E (WAVES * TILE_E) // 64 edges per block

__global__ void mace_init_out_kernel(float* __restrict__ out,
                                     const float* __restrict__ bias,
                                     int total) {
    int idx = blockIdx.x * blockDim.x + threadIdx.x;
    if (idx < total) out[idx] = bias[idx & (OUTC - 1)];
}

__launch_bounds__(128)
__global__ void mace_edge_kernel(const float* __restrict__ node_features,
                                 const float* __restrict__ edge_vectors,
                                 const long long* __restrict__ edge_index, // (2, E) int64
                                 const float* __restrict__ W,              // (512, 32)
                                 float* __restrict__ out,                  // (A, 32)
                                 int n_edges) {
    // W transposed to f16 in LDS: Wt[o][k] = W[k*32+o]. Rows are 1KB, so the
    // 16B runs of 8 halves read below are 16B-aligned.
    __shared__ __align__(32) _Float16 Wt[OUTC][KDIM];          // 32 KB
    __shared__ float Yl[WAVES][TILE_E][IRR];                   // 4 KB
    __shared__ float Fl[WAVES][TILE_E][HIDDEN];                // 8 KB
    __shared__ int   Tgt[WAVES][TILE_E];
    __shared__ int   Src[WAVES][TILE_E];

    const int tid  = threadIdx.x;
    const int lane = tid & 31;
    const int wave = tid >> 5;
    const int col  = lane & 15;   // N column / edge-in-tile for A rows
    const int hi   = lane >> 4;   // half-wave id (K sub-block selector)

    // ---- Stage W (512x32 f32 -> f16 transposed), cooperative, once per block
    for (int i = tid; i < KDIM * OUTC; i += 128) {
        int k = i >> 5;          // i / 32
        int o = i & 31;          // i % 32
        Wt[o][k] = (_Float16)W[i];
    }

    // ---- Stage per-tile edge data: SH values, src/tgt indices -------------
    const int ebase = blockIdx.x * BLOCK_E + wave * TILE_E;
    {
        // both half-waves redundantly compute edge `col` (benign dup writes,
        // keeps EXEC fully populated ahead of the WMMA section)
        int ge = ebase + col;
        bool valid = ge < n_edges;
        int gec = valid ? ge : 0;

        float x = edge_vectors[gec * 3 + 0];
        float y = edge_vectors[gec * 3 + 1];
        float z = edge_vectors[gec * 3 + 2];
        long long s = edge_index[gec];
        long long t = edge_index[(long long)n_edges + gec];

        float r = sqrtf(x * x + y * y + z * z);
        float inv = 1.0f / fmaxf(r, 1e-12f);
        x *= inv; y *= inv; z *= inv;
        float x2 = x * x, y2 = y * y, z2 = z * z;

        float Y[IRR];
        Y[0]  = 0.28209479177387814f;
        Y[1]  = 0.4886025119029199f * y;
        Y[2]  = 0.4886025119029199f * z;
        Y[3]  = 0.4886025119029199f * x;
        Y[4]  = 1.0925484305920792f * x * y;
        Y[5]  = 1.0925484305920792f * y * z;
        Y[6]  = 0.31539156525252005f * (3.0f * z2 - 1.0f);
        Y[7]  = 1.0925484305920792f * x * z;
        Y[8]  = 0.5462742152960396f * (x2 - y2);
        Y[9]  = 0.5900435899266435f * y * (3.0f * x2 - y2);
        Y[10] = 2.890611442640554f * x * y * z;
        Y[11] = 0.4570457994644658f * y * (5.0f * z2 - 1.0f);
        Y[12] = 0.3731763325901154f * z * (5.0f * z2 - 3.0f);
        Y[13] = 0.4570457994644658f * x * (5.0f * z2 - 1.0f);
        Y[14] = 1.445305721320277f  * z * (x2 - y2);
        Y[15] = 0.5900435899266435f * x * (x2 - 3.0f * y2);

        float m = valid ? 1.0f : 0.0f;
#pragma unroll
        for (int j = 0; j < IRR; ++j) Yl[wave][col][j] = Y[j] * m;
        Src[wave][col] = valid ? (int)s : 0;
        Tgt[wave][col] = valid ? (int)t : -1;
    }
    __syncthreads();

    // ---- Gather node features for the tile: 16 edges x 32 f32 -------------
    {
        int le   = lane >> 1;                 // edge 0..15
        int half = lane & 1;                  // which 16-float half
        int s = Src[wave][le];
        const float4* fp = reinterpret_cast<const float4*>(
            node_features + (long long)s * HIDDEN + half * 16);
        float4* dp = reinterpret_cast<float4*>(&Fl[wave][le][half * 16]);
#pragma unroll
        for (int q = 0; q < 4; ++q) dp[q] = fp[q];
    }
    __syncthreads();

    // ---- WMMA: P[16 edges x 32 out] = A(16x512) @ W(512x32), K-tiled -------
    // A layout (16-bit A 16x32): lane (col,hi) row M=col holds
    //   a[j]   -> K = Kb + hi*8 + j      (hidden channel h = 2*kk, irrep i = hi*8+j)
    //   a[8+j] -> K = Kb + 16 + hi*8 + j (hidden channel h = 2*kk+1, same irreps)
    _Float16 yh[8];
#pragma unroll
    for (int j = 0; j < 8; ++j) yh[j] = (_Float16)Yl[wave][col][hi * 8 + j];

    v8f c0 = {};   // out columns 0..15
    v8f c1 = {};   // out columns 16..31
#pragma unroll
    for (int kk = 0; kk < KDIM / 32; ++kk) {
        _Float16 h0 = (_Float16)Fl[wave][col][2 * kk + 0];
        _Float16 h1 = (_Float16)Fl[wave][col][2 * kk + 1];
        v16h a;
#pragma unroll
        for (int j = 0; j < 8; ++j) {
            a[j]     = h0 * yh[j];
            a[8 + j] = h1 * yh[j];
        }
        // B layout (32x16): b[j] -> K = Kb + hi*8 + j, b[8+j] -> K = Kb + 16 + hi*8 + j;
        // column N = lane%16 -> f16 from Wt[N][...]
        v16h b0, b1;
#pragma unroll
        for (int j = 0; j < 8; ++j) {
            b0[j]     = Wt[col][kk * 32 + hi * 8 + j];
            b0[8 + j] = Wt[col][kk * 32 + 16 + hi * 8 + j];
            b1[j]     = Wt[16 + col][kk * 32 + hi * 8 + j];
            b1[8 + j] = Wt[16 + col][kk * 32 + 16 + hi * 8 + j];
        }
        c0 = __builtin_amdgcn_wmma_f32_16x16x32_f16(false, a, false, b0,
                                                    (short)0, c0, false, false);
        c1 = __builtin_amdgcn_wmma_f32_16x16x32_f16(false, a, false, b1,
                                                    (short)0, c1, false, false);
    }

    // ---- Scatter: C/D layout -> lane holds rows M = hi*8 + r, col N=lane%16
#pragma unroll
    for (int r = 0; r < 8; ++r) {
        int m = hi * 8 + r;               // edge within tile
        int t = Tgt[wave][m];
        if (t >= 0) {
            atomicAdd(&out[(long long)t * OUTC + col],       c0[r]);
            atomicAdd(&out[(long long)t * OUTC + 16 + col],  c1[r]);
        }
    }
}

extern "C" void kernel_launch(void* const* d_in, const int* in_sizes, int n_in,
                              void* d_out, int out_size, void* d_ws, size_t ws_size,
                              hipStream_t stream) {
    const float*     node_features = (const float*)d_in[0];
    const float*     edge_vectors  = (const float*)d_in[1];
    const long long* edge_index    = (const long long*)d_in[2]; // int64 per reference
    const float*     W             = (const float*)d_in[3];
    const float*     bias          = (const float*)d_in[4];
    float*           out           = (float*)d_out;

    const int n_atoms = in_sizes[0] / HIDDEN;
    const int n_edges = in_sizes[1] / 3;

    const int total = n_atoms * OUTC;
    mace_init_out_kernel<<<(total + 255) / 256, 256, 0, stream>>>(out, bias, total);

    const int blocks = (n_edges + BLOCK_E - 1) / BLOCK_E;
    mace_edge_kernel<<<blocks, 128, 0, stream>>>(node_features, edge_vectors,
                                                 edge_index, W, out, n_edges);
}
